// GAT_5677946765716
// MI455X (gfx1250) — matrix operand order repaired and checked by reading, not compiled, not run on hardware
//
#include <hip/hip_runtime.h>
#include <hip/hip_bf16.h>

#define GN 6144      // nodes
#define FIN 128      // input features
#define NHID 64      // hidden per head
#define NH 4         // heads
#define NCLS 16      // classes
#define X2F (NH*NHID)  // 256

typedef _Float16 v16h __attribute__((ext_vector_type(16)));
typedef _Float16 v8h  __attribute__((ext_vector_type(8)));
typedef float    v8f  __attribute__((ext_vector_type(8)));

__device__ __forceinline__ int lane_id() { return threadIdx.x & 31; }

// A fragment 16x32 f16 from row-major MxK memory (K contiguous), leading dim ld (halves).
// Per ISA: lanes 0-15 -> row M=lane, K = {0..7, 16..23}; lanes 16-31 -> row M=lane-16, K = {8..15, 24..31}.
__device__ __forceinline__ v16h load_a16x32(const _Float16* base, int ld) {
  int l = lane_id();
  int row = l & 15, ks = l >> 4;
  const _Float16* p = base + (size_t)row * ld + ks * 8;
  v8h lo = *(const v8h*)(p);
  v8h hi = *(const v8h*)(p + 16);
  v16h a;
#pragma unroll
  for (int i = 0; i < 8; ++i) { a[i] = lo[i]; a[i + 8] = hi[i]; }
  return a;
}

// B fragment 32x16 f16 loaded from B^T stored row-major (N x K, K contiguous), ld in halves.
// Per ISA: lanes 0-15 -> col N=lane, K = {0..7, 16..23}; lanes 16-31 -> K = {8..15, 24..31}.
__device__ __forceinline__ v16h load_b32x16(const _Float16* baseT, int ld) {
  int l = lane_id();
  int n = l & 15, ks = l >> 4;
  const _Float16* p = baseT + (size_t)n * ld + ks * 8;
  v8h lo = *(const v8h*)(p);
  v8h hi = *(const v8h*)(p + 16);
  v16h b;
#pragma unroll
  for (int i = 0; i < 8; ++i) { b[i] = lo[i]; b[i + 8] = hi[i]; }
  return b;
}

__device__ __forceinline__ v8f wmma_f16(v16h a, v16h b, v8f c) {
  return __builtin_amdgcn_wmma_f32_16x16x32_f16(false, a, false, b, (short)0, c, false, false);
}

__device__ __forceinline__ float lrelu(float x) { return x >= 0.f ? x : 0.2f * x; }
__device__ __forceinline__ float eluf(float x)  { return x >  0.f ? x : __expf(x) - 1.f; }

// ---------------------------------------------------------------- converters
__global__ void k_feat_f16(const float* __restrict__ f, _Float16* __restrict__ o) {
  int i = blockIdx.x * 256 + threadIdx.x;           // GN*FIN = 786432
  o[i] = (_Float16)f[i];
}

// WeffT[h][f][k] = sum_b W0[h][b*128+k][f]  (f16, K contiguous for B loads)
__global__ void k_weffT(const float* __restrict__ W0, _Float16* __restrict__ wt) {
  int idx = blockIdx.x * 256 + threadIdx.x;         // NH*NHID*FIN = 32768
  int h = idx / (NHID * FIN);
  int r = idx % (NHID * FIN);
  int f = r / FIN, k = r % FIN;
  float s = 0.f;
#pragma unroll
  for (int b = 0; b < 4; ++b)
    s += W0[((size_t)h * 512 + b * 128 + k) * NHID + f];
  wt[((size_t)h * NHID + f) * FIN + k] = (_Float16)s;
}

// WoutT[c][k] = Wout[k][c]
__global__ void k_woutT(const float* __restrict__ W, _Float16* __restrict__ wt) {
  int idx = blockIdx.x * 256 + threadIdx.x;         // NCLS*X2F = 4096
  int c = idx / X2F, k = idx % X2F;
  wt[(size_t)c * X2F + k] = (_Float16)W[(size_t)k * NCLS + c];
}

// ------------------------------------------------- GEMM h = featF16 @ WeffT
__global__ void k_gemm_h(const _Float16* __restrict__ featF,
                         const _Float16* __restrict__ weffT,
                         float* __restrict__ hF32, _Float16* __restrict__ hT) {
  int w = blockIdx.x * 8 + (threadIdx.x >> 5);      // 6144 tiles: head x 384 x 4
  int head = w / 1536;
  int rem = w % 1536;
  int i0 = (rem >> 2) * 16;
  int f0 = (rem & 3) * 16;
  v8f acc;
#pragma unroll
  for (int r = 0; r < 8; ++r) acc[r] = 0.f;
#pragma unroll
  for (int kk = 0; kk < FIN; kk += 32) {
    v16h a = load_a16x32(featF + (size_t)i0 * FIN + kk, FIN);
    v16h b = load_b32x16(weffT + ((size_t)head * NHID + f0) * FIN + kk, FIN);
    acc = wmma_f16(a, b, acc);
  }
  int l = lane_id(), hi = l >> 4, col = l & 15;
#pragma unroll
  for (int r = 0; r < 8; ++r) {
    int rr = i0 + r + 8 * hi;
    float v = acc[r];
    hF32[((size_t)head * GN + rr) * NHID + f0 + col] = v;
    hT[(size_t)head * NHID * GN + (size_t)(f0 + col) * GN + rr] = (_Float16)v;
  }
}

// --------------------------------------------------- s/t scalars, layer 1
__global__ void k_st1(const float* __restrict__ hF32, const float* __restrict__ a1,
                      const float* __restrict__ a2, float* __restrict__ s1,
                      float* __restrict__ t1) {
  int idx = blockIdx.x * 256 + threadIdx.x;         // NH*GN = 24576
  int head = idx / GN, n = idx % GN;
  const float* row = hF32 + ((size_t)head * GN + n) * NHID;
  float ss = 0.f, tt = 0.f;
#pragma unroll 4
  for (int f = 0; f < NHID; ++f) {
    float v = row[f];
    ss += v * a1[head * NHID + f];
    tt += v * a2[head * NHID + f];
  }
  s1[idx] = ss;
  t1[idx] = tt;
}

// --------------- fused attention, layer 1: ALL 4 heads per block
// Block: 256 threads = 8 waves; one 16-row block of the N x N attention.
__global__ void k_attn1(const float* __restrict__ adj, const float* __restrict__ s1,
                        const float* __restrict__ t1, const _Float16* __restrict__ hT,
                        _Float16* __restrict__ x2) {
  int i0 = blockIdx.x * 16;
  int tid = threadIdx.x;
  int wv = tid >> 5, l = tid & 31;
  __shared__ float sm_s[NH][16], sm_m[NH][16], sm_il[NH][16];
  __shared__ float red[8][16][2];
  __shared__ __align__(32) _Float16 P[2][NH][16][32];
  if (tid < 64) sm_s[tid >> 4][tid & 15] = s1[(size_t)(tid >> 4) * GN + i0 + (tid & 15)];
  __syncthreads();

  // ---- stage 1: per-(row,head) max / sum-exp; each adj element read ONCE
  int c = tid & 63;                 // column lane (stride 64)
  int g = tid >> 6;                 // row group: rows 4g..4g+3
  float sreg[16];                   // k = h*4+rr
#pragma unroll
  for (int h = 0; h < NH; ++h)
#pragma unroll
    for (int rr = 0; rr < 4; ++rr) sreg[h * 4 + rr] = sm_s[h][4 * g + rr];
  float m[16], ls[16];
#pragma unroll
  for (int k = 0; k < 16; ++k) { m[k] = -3.0e38f; ls[k] = 0.f; }
  for (int j = c; j < GN; j += 64) {
    float tj[NH];
#pragma unroll
    for (int h = 0; h < NH; ++h) tj[h] = t1[(size_t)h * GN + j];
    const float* arow = adj + (size_t)(i0 + 4 * g) * GN + j;
    __builtin_prefetch(arow + 64, 0, 1);
#pragma unroll
    for (int rr = 0; rr < 4; ++rr) {
      float a = arow[(size_t)rr * GN];
      float msk = -1.0e10f * (1.0f - a);
#pragma unroll
      for (int h = 0; h < NH; ++h) {
        int k = h * 4 + rr;
        float lg = lrelu(sreg[k] + tj[h]) + msk;
        float mn = fmaxf(m[k], lg);
        ls[k] = ls[k] * __expf(m[k] - mn) + __expf(lg - mn);
        m[k] = mn;
      }
    }
  }
#pragma unroll
  for (int off = 16; off >= 1; off >>= 1) {
#pragma unroll
    for (int k = 0; k < 16; ++k) {
      float mo = __shfl_xor(m[k], off, 32);
      float lo = __shfl_xor(ls[k], off, 32);
      float mn = fmaxf(m[k], mo);
      ls[k] = ls[k] * __expf(m[k] - mn) + lo * __expf(mo - mn);
      m[k] = mn;
    }
  }
  if (l == 0) {
#pragma unroll
    for (int k = 0; k < 16; ++k) { red[wv][k][0] = m[k]; red[wv][k][1] = ls[k]; }
  }
  __syncthreads();
  if (tid < 64) {                   // finalize: combine the 2 waves of each row group
    int h = tid >> 4, row = tid & 15;
    int g2 = row >> 2, k = h * 4 + (row & 3);
    float m0 = red[2 * g2][k][0], l0 = red[2 * g2][k][1];
    float m1 = red[2 * g2 + 1][k][0], l1 = red[2 * g2 + 1][k][1];
    float mn = fmaxf(m0, m1);
    float ll = l0 * __expf(m0 - mn) + l1 * __expf(m1 - mn);
    sm_m[h][row] = mn;
    sm_il[h][row] = 1.0f / ll;
  }
  __syncthreads();

  // ---- stage 2: per chunk build P for all 4 heads (LDS, double-buffered),
  // wave w -> (head = w/2, f-tiles {16*(w&1), 16*(w&1)+32}), 2 WMMAs/chunk
  int head = wv >> 1;
  int f0a = (wv & 1) * 16;
  int f0b = f0a + 32;
  const _Float16* hTh = hT + (size_t)head * NHID * GN;
  v8f acc1, acc2;
#pragma unroll
  for (int r = 0; r < 8; ++r) { acc1[r] = 0.f; acc2[r] = 0.f; }
  for (int cix = 0; cix < GN / 32; ++cix) {
    int j0 = cix * 32;
    _Float16(*Pb)[16][32] = P[cix & 1];
#pragma unroll
    for (int e = 0; e < 8; ++e) {
      int idx = e * 256 + tid;      // 4h x 16i x 32j = 2048
      int h = idx >> 9;
      int rem = idx & 511;
      int i = rem >> 5, j = rem & 31;
      float a = adj[(size_t)(i0 + i) * GN + j0 + j];
      float lg = lrelu(sm_s[h][i] + t1[(size_t)h * GN + j0 + j]) - 1.0e10f * (1.0f - a);
      Pb[h][i][j] = (_Float16)(__expf(lg - sm_m[h][i]) * sm_il[h][i]);
    }
    __syncthreads();
    v16h af  = load_a16x32(&Pb[head][0][0], 32);
    v16h bf1 = load_b32x16(hTh + (size_t)f0a * GN + j0, GN);
    v16h bf2 = load_b32x16(hTh + (size_t)f0b * GN + j0, GN);
    acc1 = wmma_f16(af, bf1, acc1);
    acc2 = wmma_f16(af, bf2, acc2);
  }
  int hi = l >> 4, col = l & 15;
#pragma unroll
  for (int r = 0; r < 8; ++r) {
    int rr = i0 + r + 8 * hi;
    x2[(size_t)rr * X2F + head * NHID + f0a + col] = (_Float16)eluf(acc1[r]);
    x2[(size_t)rr * X2F + head * NHID + f0b + col] = (_Float16)eluf(acc2[r]);
  }
}

// ------------------------------------------------- GEMM h2 = x2 @ Wout
__global__ void k_gemm_h2(const _Float16* __restrict__ x2, const _Float16* __restrict__ woutT,
                          float* __restrict__ h2, _Float16* __restrict__ h2T) {
  int w = blockIdx.x * 4 + (threadIdx.x >> 5);      // 384 row blocks
  int i0 = w * 16;
  v8f acc;
#pragma unroll
  for (int r = 0; r < 8; ++r) acc[r] = 0.f;
#pragma unroll
  for (int kk = 0; kk < X2F; kk += 32) {
    v16h a = load_a16x32(x2 + (size_t)i0 * X2F + kk, X2F);
    v16h b = load_b32x16(woutT + kk, X2F);
    acc = wmma_f16(a, b, acc);
  }
  int l = lane_id(), hi = l >> 4, col = l & 15;
#pragma unroll
  for (int r = 0; r < 8; ++r) {
    int rr = i0 + r + 8 * hi;
    float v = acc[r];
    h2[(size_t)rr * NCLS + col] = v;
    h2T[(size_t)col * GN + rr] = (_Float16)v;
  }
}

__global__ void k_st2(const float* __restrict__ h2, const float* __restrict__ a1,
                      const float* __restrict__ a2, float* __restrict__ s2,
                      float* __restrict__ t2) {
  int n = blockIdx.x * 256 + threadIdx.x;           // GN
  const float* row = h2 + (size_t)n * NCLS;
  float ss = 0.f, tt = 0.f;
#pragma unroll
  for (int c = 0; c < NCLS; ++c) { ss += row[c] * a1[c]; tt += row[c] * a2[c]; }
  s2[n] = ss;
  t2[n] = tt;
}

// ------------- fused attention layer 2 + ELU + per-row softmax -> d_out
__global__ void k_attn2(const float* __restrict__ adj, const float* __restrict__ s2,
                        const float* __restrict__ t2, const _Float16* __restrict__ h2T,
                        float* __restrict__ out) {
  int i0 = blockIdx.x * 16;
  int tid = threadIdx.x;                            // 128 threads = 4 waves
  int wv = tid >> 5, l = tid & 31;
  __shared__ float sm_s[16], sm_m[16], sm_il[16];
  __shared__ float red[4][16][2];
  __shared__ __align__(32) _Float16 P[2][4][16][32];
  __shared__ float part[4][16][16];
  if (tid < 16) sm_s[tid] = s2[i0 + tid];
  __syncthreads();

  float m[16], ls[16];
#pragma unroll
  for (int r = 0; r < 16; ++r) { m[r] = -3.0e38f; ls[r] = 0.f; }
  for (int j = tid; j < GN; j += 128) {
    float tj = t2[j];
    const float* arow = adj + (size_t)i0 * GN + j;
    __builtin_prefetch(arow + 128, 0, 1);
#pragma unroll
    for (int r = 0; r < 16; ++r) {
      float a = arow[(size_t)r * GN];
      float lg = lrelu(sm_s[r] + tj) - 1.0e10f * (1.0f - a);
      float mn = fmaxf(m[r], lg);
      ls[r] = ls[r] * __expf(m[r] - mn) + __expf(lg - mn);
      m[r] = mn;
    }
  }
#pragma unroll
  for (int off = 16; off >= 1; off >>= 1) {
#pragma unroll
    for (int r = 0; r < 16; ++r) {
      float mo = __shfl_xor(m[r], off, 32);
      float lo = __shfl_xor(ls[r], off, 32);
      float mn = fmaxf(m[r], mo);
      ls[r] = ls[r] * __expf(m[r] - mn) + lo * __expf(mo - mn);
      m[r] = mn;
    }
  }
  if (l == 0) {
#pragma unroll
    for (int r = 0; r < 16; ++r) { red[wv][r][0] = m[r]; red[wv][r][1] = ls[r]; }
  }
  __syncthreads();
  if (tid < 16) {
    float mm = red[0][tid][0], ll = red[0][tid][1];
#pragma unroll
    for (int w = 1; w < 4; ++w) {
      float mo = red[w][tid][0], lo = red[w][tid][1];
      float mn = fmaxf(mm, mo);
      ll = ll * __expf(mm - mn) + lo * __expf(mo - mn);
      mm = mn;
    }
    sm_m[tid] = mm;
    sm_il[tid] = 1.0f / ll;
  }
  __syncthreads();

  // stage 2: waves split the K (column) dimension; 128 columns per super-iter
  v8f acc;
#pragma unroll
  for (int r = 0; r < 8; ++r) acc[r] = 0.f;
  for (int sc = 0; sc < GN / 128; ++sc) {
    int jbase = sc * 128;
    _Float16(*Pb)[16][32] = P[sc & 1];
#pragma unroll
    for (int e = 0; e < 16; ++e) {
      int idx = e * 128 + tid;
      int sub = idx >> 9;
      int rem = idx & 511;
      int i = rem >> 5, j = rem & 31;
      int jc = jbase + sub * 32 + j;
      float a = adj[(size_t)(i0 + i) * GN + jc];
      float lg = lrelu(sm_s[i] + t2[jc]) - 1.0e10f * (1.0f - a);
      Pb[sub][i][j] = (_Float16)(__expf(lg - sm_m[i]) * sm_il[i]);
    }
    __syncthreads();
    v16h af = load_a16x32(&Pb[wv][0][0], 32);
    v16h bf = load_b32x16(h2T + jbase + wv * 32, GN);
    acc = wmma_f16(af, bf, acc);
  }
  {
    int hi = l >> 4, col = l & 15;
#pragma unroll
    for (int r = 0; r < 8; ++r) part[wv][r + 8 * hi][col] = acc[r];
  }
  __syncthreads();
  if (tid < 16) {
    int i = tid;
    float v[NCLS];
    float mx = -3.0e38f;
#pragma unroll
    for (int cc = 0; cc < NCLS; ++cc) {
      float x = part[0][i][cc] + part[1][i][cc] + part[2][i][cc] + part[3][i][cc];
      x = eluf(x);
      v[cc] = x;
      mx = fmaxf(mx, x);
    }
    float sum = 0.f;
#pragma unroll
    for (int cc = 0; cc < NCLS; ++cc) { v[cc] = __expf(v[cc] - mx); sum += v[cc]; }
    float inv = 1.f / sum;
#pragma unroll
    for (int cc = 0; cc < NCLS; ++cc) out[(size_t)(i0 + i) * NCLS + cc] = v[cc] * inv;
  }
}

// ---------------------------------------------------------------------------
static inline size_t al256(size_t x) { return (x + 255) & ~(size_t)255; }

extern "C" void kernel_launch(void* const* d_in, const int* in_sizes, int n_in,
                              void* d_out, int out_size, void* d_ws, size_t ws_size,
                              hipStream_t stream) {
  (void)in_sizes; (void)n_in; (void)out_size; (void)ws_size;
  const float* adj   = (const float*)d_in[0];
  const float* feat  = (const float*)d_in[1];
  const float* W0    = (const float*)d_in[2];
  const float* a1_0  = (const float*)d_in[3];
  const float* a2_0  = (const float*)d_in[4];
  const float* Wout  = (const float*)d_in[5];
  const float* a1o   = (const float*)d_in[6];
  const float* a2o   = (const float*)d_in[7];
  float* out = (float*)d_out;

  char* ws = (char*)d_ws;
  size_t cur = 0;
  auto take = [&](size_t bytes) { char* p = ws + cur; cur = al256(cur + bytes); return p; };
  _Float16* featF = (_Float16*)take((size_t)GN * FIN * 2);
  _Float16* weffT = (_Float16*)take((size_t)NH * NHID * FIN * 2);
  _Float16* woutT = (_Float16*)take((size_t)NCLS * X2F * 2);
  _Float16* hT    = (_Float16*)take((size_t)NH * NHID * GN * 2);
  float*    hF32  = (float*)   take((size_t)NH * GN * NHID * 4);
  float*    s1    = (float*)   take((size_t)NH * GN * 4);
  float*    t1    = (float*)   take((size_t)NH * GN * 4);
  _Float16* x2    = (_Float16*)take((size_t)GN * X2F * 2);
  float*    h2    = (float*)   take((size_t)GN * NCLS * 4);
  _Float16* h2T   = (_Float16*)take((size_t)NCLS * GN * 2);
  float*    s2    = (float*)   take((size_t)GN * 4);
  float*    t2    = (float*)   take((size_t)GN * 4);

  k_feat_f16<<<(GN * FIN) / 256, 256, 0, stream>>>(feat, featF);
  k_weffT<<<(NH * NHID * FIN) / 256, 256, 0, stream>>>(W0, weffT);
  k_woutT<<<(NCLS * X2F) / 256, 256, 0, stream>>>(Wout, woutT);
  k_gemm_h<<<768, 256, 0, stream>>>(featF, weffT, hF32, hT);
  k_st1<<<(NH * GN) / 256, 256, 0, stream>>>(hF32, a1_0, a2_0, s1, t1);
  k_attn1<<<GN / 16, 256, 0, stream>>>(adj, s1, t1, hT, x2);
  k_gemm_h2<<<96, 128, 0, stream>>>(x2, woutT, h2, h2T);
  k_st2<<<GN / 256, 256, 0, stream>>>(h2, a1o, a2o, s2, t2);
  k_attn2<<<GN / 16, 128, 0, stream>>>(adj, s2, t2, h2T, out);
}
